// AttentionLayer_41214506172783
// MI455X (gfx1250) — matrix operand order repaired and checked by reading, not compiled, not run on hardware
//
#include <hip/hip_runtime.h>

typedef __attribute__((ext_vector_type(16))) _Float16 v16h;
typedef __attribute__((ext_vector_type(8)))  float    v8f;
typedef __attribute__((ext_vector_type(4)))  unsigned int u32x4;
typedef __attribute__((ext_vector_type(4)))  int      i32x4;
typedef __attribute__((ext_vector_type(8)))  int      i32x8;

#define B_   8192
#define S_   100
#define E_   64
#define H1_  128
#define H2_  64
#define SP_  112          // S padded to 7*16
#define NTILES 7
#define KSTRIDE 68        // keys LDS row stride (floats): 16B-aligned rows AND
                          // conflict-free column reads (bank = (4s+e) mod 64)
#define H0STR 132         // h0 LDS row stride (halves)
#define H1STR 66          // h1 LDS row stride (halves)
#define NWAVES 4

// ---------------------------------------------------------------------------
// Pack W0 [256,128] and W1 [128,64] (row-major f32) into fp16 WMMA B-fragments.
// B-matrix layout (CDNA5 ISA 7.12.2 / sparse-B table): lane n (n = lane&15)
// holds column n; half i<8 holds K=kt*32+8*(lane>>4)+i, half i>=8 holds
// K=kt*32+16+8*(lane>>4)+(i-8).
// Fragment fi = nt*KT + kt stored contiguously: fi*512 + lane*16 halves.
// ---------------------------------------------------------------------------
__global__ __launch_bounds__(32) void din_pack_weights(
    const float* __restrict__ W0, const float* __restrict__ W1,
    _Float16* __restrict__ w0p, _Float16* __restrict__ w1p)
{
  const int frag = blockIdx.x;      // 0..63 -> W0, 64..79 -> W1
  const int lane = threadIdx.x;     // 32 lanes
  const float* W; _Float16* dst; int fi, nt, kt, N;
  if (frag < 64) { fi = frag;      nt = fi >> 3; kt = fi & 7; N = H1_; W = W0; dst = w0p; }
  else           { fi = frag - 64; nt = fi >> 2; kt = fi & 3; N = H2_; W = W1; dst = w1p; }
  const int n     = nt * 16 + (lane & 15);
  const int kbase = kt * 32 + ((lane >> 4) << 3);
  _Float16* o = dst + (size_t)fi * 512 + lane * 16;
#pragma unroll
  for (int i = 0; i < 16; ++i)
    o[i] = (_Float16)W[(size_t)(kbase + (i & 7) + ((i >> 3) << 4)) * N + n];
}

// ---------------------------------------------------------------------------
// One workgroup (4 wave32) per batch element b.
// __launch_bounds__(128, 4): >=4 waves/SIMD (LDS limits to ~5 blocks/WGP).
// nt loops deliberately NOT unrolled (full unroll -> 512 VGPRs of hoisted
// B-fragments -> scratch spills in front of every WMMA).
// Keys tile is staged via the Tensor Data Mover: one 2D-tile descriptor,
// 100x64 f32, with TDM LDS-padding (4 DWORDs after every 64) producing the
// conflict-free KSTRIDE=68 layout directly.
// ---------------------------------------------------------------------------
__global__ __launch_bounds__(128, 4) void din_attention(
    const float* __restrict__ query,
    const float* __restrict__ keys,
    const int*   __restrict__ keys_length,
    const _Float16* __restrict__ w0p,
    const float* __restrict__ b0,
    const _Float16* __restrict__ w1p,
    const float* __restrict__ b1,
    const float* __restrict__ W2,
    const float* __restrict__ b2,
    float* __restrict__ outp,
    float* __restrict__ attnp)
{
  __shared__ __align__(16) float qs[E_];
  __shared__ __align__(16) float ks[SP_ * KSTRIDE];   // 30464 B, fp32 keys (pad rows zeroed)
  __shared__ __align__(16) float w2s[H2_];
  __shared__ float sc[SP_];
  __shared__ float at[SP_];
  __shared__ __align__(16) _Float16 h0s[NWAVES][16 * H0STR];  // per-wave transient tiles
  __shared__ __align__(16) _Float16 h1s[NWAVES][16 * H1STR];

  const int b    = blockIdx.x;
  const int tid  = threadIdx.x;
  const int wave = tid >> 5, lane = tid & 31;
  const int lrow = lane & 15, lhi = lane >> 4;

  // ---- stage q, W2 (VALU) ----
  if (tid < 16) {
    *(float4*)&qs[tid * 4]  = ((const float4*)(query + (size_t)b * E_))[tid];
    *(float4*)&w2s[tid * 4] = ((const float4*)W2)[tid];
  }
  // ---- zero the 12 padded key rows (disjoint from the TDM-written region) ----
  {
    const float4 z = make_float4(0.f, 0.f, 0.f, 0.f);
    for (int i = tid; i < 12 * (E_ / 4); i += 128) {
      const int s = S_ + (i >> 4), e4 = i & 15;
      *(float4*)&ks[s * KSTRIDE + e4 * 4] = z;
    }
  }
#if __has_builtin(__builtin_amdgcn_tensor_load_to_lds)
  // ---- keys[b] (100x64 f32) -> LDS via Tensor Data Mover ----
  if (wave == 0) {                      // wave-uniform; TDM is a wave-level op
    const unsigned long long ga =
        (unsigned long long)(const void*)(keys + (size_t)b * S_ * E_);
    // D# group 0: count=1 | lds_addr | global_addr[56:0] | type=2 (ISA 8.3)
    u32x4 g0;
    g0.x = 1u;
    g0.y = (unsigned int)(size_t)(void*)&ks[0];     // flat low 32 = LDS offset
    g0.z = (unsigned int)ga;
    g0.w = ((unsigned int)(ga >> 32) & 0x1FFFFFFu) | (2u << 30);
    // D# group 1 (ISA 8.4): data_size=2 (4B) | pad_enable | pad_interval=5
    // (64 DWORDs) | pad_amount=3 (4 DWORDs); tensor_dim0=64, tensor_dim1=100,
    // tile_dim0=64, tile_dim1=100, tensor_dim0_stride=64.
    i32x8 g1;
    g1[0] = (2 << 16) | (1 << 20) | (5 << 22) | (3 << 25);
    g1[1] = (E_ << 16);        // tensor_dim0[15:0] in bits [79:64->31:16]
    g1[2] = (S_ << 16);        // tensor_dim0 hi=0 | tensor_dim1[15:0]
    g1[3] = (E_ << 16);        // tensor_dim1 hi=0 | tile_dim0
    g1[4] = S_;                // tile_dim1 | tile_dim2=0
    g1[5] = E_;                // tensor_dim0_stride[31:0]
    g1[6] = 0;                 // stride hi | tensor_dim1_stride lo (unused, 2D)
    g1[7] = 0;
    i32x4 gz = {0, 0, 0, 0};   // groups 2/3 unused for 2D tile (tile_dim2=0)
#if defined(__clang_major__) && (__clang_major__ >= 23)
    i32x8 gz8 = {0, 0, 0, 0, 0, 0, 0, 0};
    __builtin_amdgcn_tensor_load_to_lds(g0, g1, gz, gz, gz8, 0);
#else
    __builtin_amdgcn_tensor_load_to_lds(g0, g1, gz, gz, 0);
#endif
    __builtin_amdgcn_s_wait_tensorcnt((short)0);    // TENSORcnt==0 before barrier
  }
#else
  // ---- fallback: vectorized b128 staging ----
  {
    const float4* kg = (const float4*)(keys + (size_t)b * S_ * E_);
    for (int i = tid; i < S_ * (E_ / 4); i += 128) {
      const int s = i >> 4, e4 = i & 15;
      *(float4*)&ks[s * KSTRIDE + e4 * 4] = kg[s * 16 + e4];
    }
  }
#endif
  __syncthreads();

  for (int tile = wave; tile < NTILES; tile += NWAVES) {   // wave-uniform -> EXEC all-ones
    const int srow = tile * 16 + lrow;

    // ================= layer 0: feats[16x256] @ W0[256x128] =================
    // A 16x32 f16 layout: lane (l&15)=row; klocal = (v>>2)*16 + lhi*8 + (v&3)*2 + h
    v16h a0[8];
#pragma unroll
    for (int kt = 0; kt < 8; ++kt) {
      const int type  = kt >> 1;          // 0:q 1:k 2:q*k 3:q-k (64-aligned blocks)
      const int ebase = (kt & 1) << 5;
      v16h a;
#pragma unroll
      for (int v = 0; v < 8; ++v) {
#pragma unroll
        for (int h = 0; h < 2; ++h) {
          const int kl = ((v >> 2) << 4) + (lhi << 3) + ((v & 3) << 1) + h;
          const int e  = ebase + kl;
          const float qv = qs[e];
          const float kv = ks[srow * KSTRIDE + e];
          const float f  = (type == 0) ? qv : (type == 1) ? kv
                         : (type == 2) ? qv * kv : qv - kv;
          a[2 * v + h] = (_Float16)f;
        }
      }
      a0[kt] = a;
    }
#pragma unroll 1
    for (int nt = 0; nt < 8; ++nt) {
      v8f c = {};
#pragma unroll
      for (int kt = 0; kt < 8; ++kt) {
        v16h bf = *(const v16h*)(w0p + (size_t)(nt * 8 + kt) * 512 + lane * 16);
        c = __builtin_amdgcn_wmma_f32_16x16x32_f16(false, a0[kt], false, bf,
                                                   (short)0, c, false, false);
      }
      const float bias = b0[nt * 16 + lrow];
#pragma unroll
      for (int r = 0; r < 8; ++r) {       // D: lane=col, rows r / r+8
        float v = c[r] + bias;
        v = v > 0.f ? v : 0.f;
        h0s[wave][(r + (lhi << 3)) * H0STR + nt * 16 + lrow] = (_Float16)v;
      }
    }

    // ================= layer 1: h0[16x128] @ W1[128x64] =================
    v16h a1[4];
#pragma unroll
    for (int kt = 0; kt < 4; ++kt) {
      v16h a;
#pragma unroll
      for (int v = 0; v < 8; ++v) {
#pragma unroll
        for (int h = 0; h < 2; ++h) {
          const int kl = ((v >> 2) << 4) + (lhi << 3) + ((v & 3) << 1) + h;
          a[2 * v + h] = h0s[wave][lrow * H0STR + kt * 32 + kl];
        }
      }
      a1[kt] = a;
    }
#pragma unroll 1
    for (int nt = 0; nt < 4; ++nt) {
      v8f c = {};
#pragma unroll
      for (int kt = 0; kt < 4; ++kt) {
        v16h bf = *(const v16h*)(w1p + (size_t)(nt * 4 + kt) * 512 + lane * 16);
        c = __builtin_amdgcn_wmma_f32_16x16x32_f16(false, a1[kt], false, bf,
                                                   (short)0, c, false, false);
      }
      const float bias = b1[nt * 16 + lrow];
#pragma unroll
      for (int r = 0; r < 8; ++r) {
        float v = c[r] + bias;
        v = v > 0.f ? v : 0.f;
        h1s[wave][(r + (lhi << 3)) * H1STR + nt * 16 + lrow] = (_Float16)v;
      }
    }

    // ================= layer 2: scores = h1 @ W2 + b2 =================
    float acc = 0.f;
#pragma unroll
    for (int i = 0; i < 32; ++i) {
      const int e = (lhi << 5) + i;
      acc += (float)h1s[wave][lrow * H1STR + e] * w2s[e];
    }
    acc += __shfl_xor(acc, 16);
    if (lhi == 0) sc[tile * 16 + lrow] = acc + b2[0];
  }
  __syncthreads();

  // ================= masked softmax over S (wave 0) =================
  if (wave == 0) {
    const int len = keys_length[b];                 // >= 1
    float m = -1e30f;
    for (int s2 = lane; s2 < len; s2 += 32) m = fmaxf(m, sc[s2]);
#pragma unroll
    for (int off = 16; off > 0; off >>= 1) m = fmaxf(m, __shfl_xor(m, off));
    float sum = 0.f;
    for (int s2 = lane; s2 < SP_; s2 += 32) {
      const float p = (s2 < len) ? __expf(sc[s2] - m) : 0.f;
      at[s2] = p;
      sum += p;
    }
#pragma unroll
    for (int off = 16; off > 0; off >>= 1) sum += __shfl_xor(sum, off);
    const float inv = 1.f / sum;
    for (int s2 = lane; s2 < S_; s2 += 32) {
      const float p = at[s2] * inv;
      at[s2] = p;
      attnp[(size_t)b * S_ + s2] = p;
    }
  }
  __syncthreads();

  // ================= out[b] = attn @ keys =================
  if (tid < E_) {
    float acc = 0.f;
    for (int s2 = 0; s2 < S_; ++s2)
      acc += at[s2] * ks[s2 * KSTRIDE + tid];
    outp[(size_t)b * E_ + tid] = acc;
  }
}

// ---------------------------------------------------------------------------
extern "C" void kernel_launch(void* const* d_in, const int* in_sizes, int n_in,
                              void* d_out, int out_size, void* d_ws, size_t ws_size,
                              hipStream_t stream) {
  const float* query = (const float*)d_in[0];
  const float* keys  = (const float*)d_in[1];
  const int*   klen  = (const int*)  d_in[2];
  const float* W0    = (const float*)d_in[3];
  const float* b0    = (const float*)d_in[4];
  const float* W1    = (const float*)d_in[5];
  const float* b1    = (const float*)d_in[6];
  const float* W2    = (const float*)d_in[7];
  const float* b2    = (const float*)d_in[8];

  _Float16* w0p = (_Float16*)d_ws;          // 64 frags * 512 halves = 64 KB
  _Float16* w1p = w0p + 64 * 512;           // 16 frags * 512 halves = 16 KB

  din_pack_weights<<<80, 32, 0, stream>>>(W0, W1, w0p, w1p);

  float* outp  = (float*)d_out;                       // [B, E]
  float* attnp = outp + (size_t)B_ * E_;              // [B, S]
  din_attention<<<B_, 128, 0, stream>>>(query, keys, klen, w0p, b0, w1p, b1,
                                        W2, b2, outp, attnp);
}
